// Node_embedding_pendulum_88656714925209
// MI455X (gfx1250) — matrix operand order repaired and checked by reading, not compiled, not run on hardware
//
#include <hip/hip_runtime.h>

// ---------------------------------------------------------------------------
// MI455X (gfx1250) implementation.
// Batched tiny-MLP workload -> mapped onto v_wmma_f32_16x16x32_f16.
// 16 rows per wave32 tile; weights pre-swizzled into CDNA5 B-fragment layout
// in LDS; activations staged through LDS for C/D-layout -> A-layout reshape.
// ---------------------------------------------------------------------------

typedef __attribute__((ext_vector_type(16))) _Float16 v16h;
typedef __attribute__((ext_vector_type(8)))  _Float16 v8h;
typedef __attribute__((ext_vector_type(8)))  float    v8f;
typedef __attribute__((ext_vector_type(4)))  float    v4f;
typedef __attribute__((ext_vector_type(2)))  float    v2f;

#define WAVES_PER_BLOCK 8
#define TPB 256

#define HID     192
#define NSC     64
#define ROT_IN  32
#define KT_HID  6     // 192/32 K-steps for layer 2
#define NT_HID  12    // 192/16 N-tiles  for layer 1
#define NT_OUT  4     // 64/16  N-tiles  for layer 2

// ---- LDS layout (bytes) ----
#define OFF_WS1   0            // 12 frags * 1024B
#define OFF_WS2   12288        // 24 frags
#define OFF_WR1   36864        // 12 frags
#define OFF_WR2   49152        // 24 frags
#define OFF_BIAS  73728        // 512 f32
#define B_BS1 0
#define B_BS2 192
#define B_BR1 256
#define B_BR2 448
#define OFF_STAGE 75776
// per-wave staging (bank-conflict padded, all offsets 16B aligned)
#define AIN_STRIDE 40          // halves per row (80B -> distinct banks, 16B mult)
#define HID_STRIDE 200         // halves per row (400B)
#define OUT_STRIDE 68          // floats per row (272B)
#define ST_AINS 0              // 16*40*2  = 1280B  (scalar MLP input, 16x32 f16)
#define ST_AINR 1280           // 1280B             (rot MLP input)
#define ST_HID  2560           // 16*200*2 = 6400B  (hidden 16x192 f16)
#define ST_OUT  8960           // 16*68*4  = 4352B  (layer-2 out, f32)
#define STAGE_BYTES 13312
#define LDS_BYTES (OFF_STAGE + WAVES_PER_BLOCK * STAGE_BYTES)   // 182272 B

// ---------------------------------------------------------------------------
// Fragment loaders.
// B-fragment (32x16 f16, pre-swizzled in LDS): lane holds v16h[i] = W[K][N]
// with K = i + 16*(lane>>4), N = ntile*16 + (lane&15). Stored contiguously:
// frag*1024 + lane*32 bytes -> two ds_load_b128 per lane.
// A-fragment (16x32 f16, row-major staging): lane (row=lane&15, g=lane>>4)
// needs K = g*16..g*16+15 -> two ds_load_b128.
// ---------------------------------------------------------------------------
__device__ __forceinline__ v16h ld_fragB(const _Float16* frags, int frag, int lane) {
  const v8h* p = (const v8h*)(frags + frag * 512 + lane * 16);
  v8h lo = p[0], hi = p[1];
  return __builtin_shufflevector(lo, hi, 0,1,2,3,4,5,6,7,8,9,10,11,12,13,14,15);
}

__device__ __forceinline__ v16h ld_fragA(const _Float16* rowbase, int strideH, int lane) {
  int row = lane & 15, g = lane >> 4;
  const v8h* p0 = (const v8h*)(rowbase + row * strideH + g * 16);
  const v8h* p1 = (const v8h*)(rowbase + row * strideH + g * 16 + 8);
  return __builtin_shufflevector(p0[0], p1[0], 0,1,2,3,4,5,6,7,8,9,10,11,12,13,14,15);
}

__device__ __forceinline__ v8f wmma_f16(v16h a, v16h b, v8f c) {
  return __builtin_amdgcn_wmma_f32_16x16x32_f16(false, a, false, b, (short)0, c,
                                                false, false);
}

// Convert one f32 weight matrix [Krows x Ncols] into pre-swizzled f16
// B-fragments in LDS (frag index = nt*ktiles + kt; K padded with zeros).
__device__ __forceinline__ void stage_weight(const float* __restrict__ W,
                                             int Krows, int Ncols, int ktiles,
                                             int ntiles, _Float16* dst, int tid) {
  int total = ktiles * ntiles * 512;
  for (int e = tid; e < total; e += TPB) {
    int frag = e >> 9, w = e & 511;
    int lane = w >> 4, i = w & 15;
    int nt = frag / ktiles, kt = frag - nt * ktiles;
    int K  = kt * 32 + i + ((lane >> 4) << 4);
    int Nc = nt * 16 + (lane & 15);
    float val = (K < Krows) ? W[K * Ncols + Nc] : 0.f;
    dst[frag * 512 + lane * 16 + i] = (_Float16)val;
  }
}

// One 2-layer MLP (K<=32 -> 192 relu -> 64) for a 16-row tile.
__device__ __forceinline__ void run_mlp(const _Float16* ainBase,
                                        const _Float16* w1f, const _Float16* w2f,
                                        const float* bias1, const float* bias2,
                                        _Float16* hid, float* outs, int lane) {
  int cl = lane & 15, g = lane >> 4;
  v16h a = ld_fragA(ainBase, AIN_STRIDE, lane);
  // ---- layer 1: 12 independent N-tiles, K=32 (one WMMA each) ----
#pragma unroll
  for (int nt = 0; nt < NT_HID; nt++) {
    float b = bias1[nt * 16 + cl];
    v8f acc = {b, b, b, b, b, b, b, b};
    acc = wmma_f16(a, ld_fragB(w1f, nt, lane), acc);
    int col = nt * 16 + cl;
#pragma unroll
    for (int m = 0; m < 8; m++) {   // fused bias+ReLU+cvt, C/D layout -> LDS
      float x = acc[m];
      hid[(m + 8 * g) * HID_STRIDE + col] = (_Float16)(x > 0.f ? x : 0.f);
    }
  }
  asm volatile("s_wait_dscnt 0" ::: "memory");
  // ---- layer 2: K=192 (6 steps) x 4 N-tiles, 4 independent acc chains ----
  v16h hA[KT_HID];
#pragma unroll
  for (int kt = 0; kt < KT_HID; kt++)
    hA[kt] = ld_fragA(hid + kt * 32, HID_STRIDE, lane);
  v8f acc2[NT_OUT];
#pragma unroll
  for (int nt = 0; nt < NT_OUT; nt++) {
    float b = bias2[nt * 16 + cl];
    acc2[nt] = (v8f){b, b, b, b, b, b, b, b};
  }
#pragma unroll
  for (int kt = 0; kt < KT_HID; kt++)
#pragma unroll
    for (int nt = 0; nt < NT_OUT; nt++)
      acc2[nt] = wmma_f16(hA[kt], ld_fragB(w2f, nt * KT_HID + kt, lane), acc2[nt]);
#pragma unroll
  for (int nt = 0; nt < NT_OUT; nt++)
#pragma unroll
    for (int m = 0; m < 8; m++)
      outs[(m + 8 * g) * OUT_STRIDE + nt * 16 + cl] = acc2[nt][m];
  asm volatile("s_wait_dscnt 0" ::: "memory");
}

__global__ void __launch_bounds__(TPB, 1)
node_embed_kernel(const float* __restrict__ v, const float* __restrict__ force,
                  const int* __restrict__ is_moving,
                  const float* __restrict__ rot_theta,
                  const float* __restrict__ Ws1, const float* __restrict__ bs1,
                  const float* __restrict__ Ws2, const float* __restrict__ bs2,
                  const float* __restrict__ Wr1, const float* __restrict__ br1,
                  const float* __restrict__ Wr2, const float* __restrict__ br2,
                  float* __restrict__ out_scalar, float* __restrict__ out_rot,
                  int n_tiles) {
  extern __shared__ __align__(16) char smem[];
  _Float16* ws1f = (_Float16*)(smem + OFF_WS1);
  _Float16* ws2f = (_Float16*)(smem + OFF_WS2);
  _Float16* wr1f = (_Float16*)(smem + OFF_WR1);
  _Float16* wr2f = (_Float16*)(smem + OFF_WR2);
  float*    bias = (float*)(smem + OFF_BIAS);

  int tid = threadIdx.x;
  // one-time cooperative weight conversion + swizzle into LDS (f32 -> f16)
  stage_weight(Ws1, 4,      HID, 1,      NT_HID, ws1f, tid);
  stage_weight(Ws2, HID,    NSC, KT_HID, NT_OUT, ws2f, tid);
  stage_weight(Wr1, ROT_IN, HID, 1,      NT_HID, wr1f, tid);
  stage_weight(Wr2, HID,    NSC, KT_HID, NT_OUT, wr2f, tid);
  for (int i = tid; i < 512; i += TPB) {
    float b;
    if (i < 192)      b = bs1[i];
    else if (i < 256) b = bs2[i - 192];
    else if (i < 448) b = br1[i - 256];
    else              b = br2[i - 448];
    bias[i] = b;
  }
  __syncthreads();

  int wave = tid >> 5, lane = tid & 31;
  char* stg = smem + OFF_STAGE + wave * STAGE_BYTES;
  _Float16* ainS = (_Float16*)(stg + ST_AINS);
  _Float16* ainR = (_Float16*)(stg + ST_AINR);
  _Float16* hid  = (_Float16*)(stg + ST_HID);
  float*    outs = (float*)(stg + ST_OUT);

  for (long long tile = (long long)blockIdx.x * WAVES_PER_BLOCK + wave;
       tile < n_tiles; tile += (long long)gridDim.x * WAVES_PER_BLOCK) {
    long long rowBase = tile * 16;
    long long nextTile = tile + (long long)gridDim.x * WAVES_PER_BLOCK;
    if (nextTile < n_tiles)
      __builtin_prefetch(rot_theta + (nextTile * 16 + (lane & 15)) * 32, 0, 0);

    // ================= per-row feature phase (lanes 0..15, one row each) ===
    if (lane < 16) {
      long long r = rowBase + lane;
      v2f fv = *(const v2f*)(force + r * 2);
      v2f vv = *(const v2f*)(v + r * 2);
      float fn = sqrtf(fv.x * fv.x + fv.y * fv.y);
      float vn = sqrtf(vv.x * vv.x + vv.y * vv.y);
      int mv = is_moving[r];
      float oh0 = (mv == 0) ? 1.f : 0.f;
      float oh1 = (mv == 1) ? 1.f : 0.f;
      // scalar MLP A row: [|f|, |v|, oh0, oh1, 0...0]  (K padded 4 -> 32)
      v8h z = {0, 0, 0, 0, 0, 0, 0, 0};
      v8h r0 = {(_Float16)fn, (_Float16)vn, (_Float16)oh0, (_Float16)oh1, 0, 0, 0, 0};
      v8h* arow = (v8h*)(ainS + lane * AIN_STRIDE);
      arow[0] = r0; arow[1] = z; arow[2] = z; arow[3] = z;

      // Fourier features cos(k*theta), sin(k*theta) via complex rotation
      // recurrence (cos(theta_f)=fx/|f|): zero transcendentals.
      float cf1 = fv.x / fn, sf1 = fv.y / fn;
      float cv1 = vv.x / vn, sv1 = vv.y / vn;
      v4f R[8];
      const v4f* Rg = (const v4f*)(rot_theta + r * 32);
#pragma unroll
      for (int k = 0; k < 8; k++) R[k] = Rg[k];   // R[k] = {R00,R01,R10,R11}

      float xin[32];
      float c = cf1, s = sf1;
#pragma unroll
      for (int k = 0; k < 8; k++) {               // x = feat @ R^T per (k)
        xin[k * 2 + 0] = c * R[k].x + s * R[k].y;
        xin[k * 2 + 1] = c * R[k].z + s * R[k].w;
        float cn = c * cf1 - s * sf1, sn = s * cf1 + c * sf1;
        c = cn; s = sn;
      }
      c = cv1; s = sv1;
#pragma unroll
      for (int k = 0; k < 8; k++) {
        xin[16 + k * 2 + 0] = c * R[k].x + s * R[k].y;
        xin[16 + k * 2 + 1] = c * R[k].z + s * R[k].w;
        float cn = c * cv1 - s * sv1, sn = s * cv1 + c * sv1;
        c = cn; s = sn;
      }
      _Float16* rrow = ainR + lane * AIN_STRIDE;
#pragma unroll
      for (int i = 0; i < 32; i++) rrow[i] = (_Float16)xin[i];
    }
    asm volatile("s_wait_dscnt 0" ::: "memory");

    // ================= scalar MLP (4 -> 192 -> 64) =========================
    run_mlp(ainS, ws1f, ws2f, bias + B_BS1, bias + B_BS2, hid, outs, lane);
    {   // coalesced copy: 16 rows x 64 f32 contiguous in d_out
      float* gdst = out_scalar + rowBase * 64;
#pragma unroll
      for (int rr = 0; rr < 16; rr++) {
        v2f val = *(const v2f*)(outs + rr * OUT_STRIDE + lane * 2);
        *(v2f*)(gdst + rr * 64 + lane * 2) = val;
      }
    }

    // ================= rot MLP (32 -> 192 -> 64) ===========================
    run_mlp(ainR, wr1f, wr2f, bias + B_BR1, bias + B_BR2, hid, outs, lane);
    // apply per-row forward rotation rot_theta and store (lanes 0..15)
    if (lane < 16) {
      long long r = rowBase + lane;
      v4f R[8];
      const v4f* Rg = (const v4f*)(rot_theta + r * 32);
#pragma unroll
      for (int k = 0; k < 8; k++) R[k] = Rg[k];
      float* gout = out_rot + r * 64;
#pragma unroll
      for (int j = 0; j < 4; j++) {
        const v4f* xs = (const v4f*)(outs + lane * OUT_STRIDE + j * 16);
        v4f* gq = (v4f*)(gout + j * 16);
#pragma unroll
        for (int q = 0; q < 4; q++) {   // pairs k=2q, 2q+1 ; out = x @ R[k]
          v4f xv = xs[q];
          v4f Ra = R[2 * q], Rb = R[2 * q + 1];
          v4f o;
          o.x = xv.x * Ra.x + xv.y * Ra.z;
          o.y = xv.x * Ra.y + xv.y * Ra.w;
          o.z = xv.z * Rb.x + xv.w * Rb.z;
          o.w = xv.z * Rb.y + xv.w * Rb.w;
          gq[q] = o;
        }
      }
    }
  }
}

extern "C" void kernel_launch(void* const* d_in, const int* in_sizes, int n_in,
                              void* d_out, int out_size, void* d_ws, size_t ws_size,
                              hipStream_t stream) {
  const float* v         = (const float*)d_in[0];
  const float* force     = (const float*)d_in[1];
  const int*   is_moving = (const int*)d_in[2];
  const float* rot_theta = (const float*)d_in[3];
  const float* Ws1 = (const float*)d_in[4];
  const float* bs1 = (const float*)d_in[5];
  const float* Ws2 = (const float*)d_in[6];
  const float* bs2 = (const float*)d_in[7];
  const float* Wr1 = (const float*)d_in[8];
  const float* br1 = (const float*)d_in[9];
  const float* Wr2 = (const float*)d_in[10];
  const float* br2 = (const float*)d_in[11];

  int n = in_sizes[0] / 2;            // N rows (1e6; divisible by 16)
  int n_tiles = n / 16;
  float* out_scalar = (float*)d_out;
  float* out_rot    = out_scalar + (long long)n * 64;

  (void)hipFuncSetAttribute((const void*)node_embed_kernel,
                            hipFuncAttributeMaxDynamicSharedMemorySize,
                            LDS_BYTES);
  int blocks = 1024;                  // ~7.6 tiles per wave; amortizes LDS init
  node_embed_kernel<<<blocks, TPB, LDS_BYTES, stream>>>(
      v, force, is_moving, rot_theta, Ws1, bs1, Ws2, bs2, Wr1, br1, Wr2, br2,
      out_scalar, out_rot, n_tiles);
}
